// ktMaxDiffuCl_20194936226533
// MI455X (gfx1250) — matrix operand order repaired and checked by reading, not compiled, not run on hardware
//
#include <hip/hip_runtime.h>
#include <hip/hip_bf16.h>

typedef __attribute__((ext_vector_type(16))) _Float16 v16h;
typedef __attribute__((ext_vector_type(8)))  _Float16 v8h;
typedef __attribute__((ext_vector_type(8)))  float    v8f;

#define NB    512
#define NS    64
#define NKN   8
#define ND    16
#define NSTEP 100
#define NUNIT 128

// packed (fragment-ordered) weight offsets in f16 workspace region, in halves
#define OFF_AGG  0       // 1 frag  * 32 lanes * 16 = 512   (K padded 16->32)
#define OFF_W1   512     // 8 nt    * 32 * 16      = 4096   (K padded 16->32)
#define OFF_W2   4608    // 8 nt*4kc* 32 * 16      = 16384
#define OFF_W3   20992   // 16384
#define OFF_W4   37376   // 4 kc    * 32 * 16      = 2048
#define PACK_TOT 39424

static __device__ __forceinline__ v8f wmma16(v16h a, v16h b, v8f c) {
  // D = A(16x32 f16) * B(32x16 f16) + C(16x16 f32)
  return __builtin_amdgcn_wmma_f32_16x16x32_f16(false, a, false, b, (short)0, c, false, false);
}

static __device__ __forceinline__ v16h cat16(v8h lo, v8h hi) {
  v16h r;
#pragma unroll
  for (int i = 0; i < 8; ++i) { r[i] = lo[i]; r[i + 8] = hi[i]; }
  return r;
}

static __device__ __forceinline__ v8h zero8() {
  v8h z;
#pragma unroll
  for (int i = 0; i < 8; ++i) z[i] = (_Float16)0.0f;
  return z;
}

// ---------------- diffusion schedule: abar_sqrt / ombar_sqrt ----------------
__global__ void sched_kernel(float* __restrict__ ab, float* __restrict__ ob) {
  int i = threadIdx.x;
  if (i < NSTEP) {
    float prod = 1.0f;
    for (int j = 0; j <= i; ++j) {
      float x = -6.0f + (12.0f / 99.0f) * (float)j;
      float beta = (1.0f / (1.0f + expf(-x))) * (0.005f - 1e-5f) + 1e-5f;
      prod *= (1.0f - beta);
    }
    ab[i] = sqrtf(prod);
    ob[i] = sqrtf(1.0f - prod);
  }
}

// ---- weight f32 -> f16 conversion, packed into WMMA B-fragment order ----
// Per fragment: 32 lanes x 16 halves, lane-contiguous (one v16h load per lane).
// B frag element e on lane (sub,ln): logical K row = kbase + (e&7) + sub*8 + (e>>3)*16, col = ln.
__global__ void prep_kernel(const float* __restrict__ aggW,
                            const float* __restrict__ W1, const float* __restrict__ W2,
                            const float* __restrict__ W3, const float* __restrict__ W4,
                            _Float16* __restrict__ h) {
  int i = blockIdx.x * 256 + threadIdx.x;
  if (i >= PACK_TOT) return;
  int idx, lane, e, sub, ln, kk;
  if (i < OFF_W1) {                       // agg_W [16,16], K padded to 32
    idx = i - OFF_AGG;
    lane = idx >> 4; e = idx & 15; sub = lane >> 4; ln = lane & 15;
    kk = (e & 7) + sub * 8 + (e >> 3) * 16;
    h[i] = (kk < 16) ? (_Float16)aggW[kk * 16 + ln] : (_Float16)0.0f;
  } else if (i < OFF_W2) {                // W1 [16,128], K padded to 32
    idx = i - OFF_W1;
    int nt = idx >> 9; int rem = idx & 511;
    lane = rem >> 4; e = rem & 15; sub = lane >> 4; ln = lane & 15;
    kk = (e & 7) + sub * 8 + (e >> 3) * 16;
    h[i] = (kk < 16) ? (_Float16)W1[kk * NUNIT + nt * 16 + ln] : (_Float16)0.0f;
  } else if (i < OFF_W3) {                // W2 [128,128]
    idx = i - OFF_W2;
    int ntkc = idx >> 9; int nt = ntkc >> 2, kc = ntkc & 3;
    int rem = idx & 511;
    lane = rem >> 4; e = rem & 15; sub = lane >> 4; ln = lane & 15;
    kk = kc * 32 + (e & 7) + sub * 8 + (e >> 3) * 16;
    h[i] = (_Float16)W2[kk * NUNIT + nt * 16 + ln];
  } else if (i < OFF_W4) {                // W3 [128,128]
    idx = i - OFF_W3;
    int ntkc = idx >> 9; int nt = ntkc >> 2, kc = ntkc & 3;
    int rem = idx & 511;
    lane = rem >> 4; e = rem & 15; sub = lane >> 4; ln = lane & 15;
    kk = kc * 32 + (e & 7) + sub * 8 + (e >> 3) * 16;
    h[i] = (_Float16)W3[kk * NUNIT + nt * 16 + ln];
  } else {                                // W4 [128,16]
    idx = i - OFF_W4;
    int kc = idx >> 9;
    int rem = idx & 511;
    lane = rem >> 4; e = rem & 15; sub = lane >> 4; ln = lane & 15;
    kk = kc * 32 + (e & 7) + sub * 8 + (e >> 3) * 16;
    h[i] = (_Float16)W4[kk * ND + ln];
  }
}

// ---------------- diffusion MLP branch: writes d_out[:, 1:17] ----------------
__global__ void __launch_bounds__(128)
diff_kernel(const float* __restrict__ usr_emb, const float* __restrict__ noise_e,
            const int* __restrict__ u, const int* __restrict__ t,
            const float* __restrict__ b1, const float* __restrict__ b2,
            const float* __restrict__ b3, const float* __restrict__ b4,
            const float* __restrict__ se1, const float* __restrict__ se2,
            const float* __restrict__ se3,
            const _Float16* __restrict__ hp,   // packed weights base
            const float* __restrict__ ab, const float* __restrict__ ob,
            float* __restrict__ out) {
  __shared__ _Float16 sX[16 * 16];
  __shared__ _Float16 sHa[16 * NUNIT];
  __shared__ _Float16 sHb[16 * NUNIT];
  __shared__ int st[16];

  const int tid  = threadIdx.x;
  const int wave = tid >> 5;
  const int lane = tid & 31;
  const int sub  = lane >> 4;
  const int ln   = lane & 15;
  const int row0 = blockIdx.x * 16;

  if (tid < 16) st[tid] = t[row0 + tid];
  for (int idx = tid; idx < 256; idx += 128) {
    int mm = idx >> 4, dd = idx & 15;
    int gm = row0 + mm;
    int tv = t[gm];
    float x = usr_emb[(long)u[gm] * ND + dd] * ab[tv] + noise_e[gm * ND + dd] * ob[tv];
    sX[idx] = (_Float16)x;
  }
  __syncthreads();

  // ---- layer 1: [16,16] @ W1[16,128]  (K padded 16->32) ----
  for (int nt = wave; nt < 8; nt += 4) {
    v8h lo = *(const v8h*)&sX[ln * 16 + sub * 8];
    v16h af = cat16(lo, zero8());
    v16h bf = *(const v16h*)(hp + OFF_W1 + (nt * 32 + lane) * 16);
    v8f c = {0.f, 0.f, 0.f, 0.f, 0.f, 0.f, 0.f, 0.f};
    c = wmma16(af, bf, c);
#pragma unroll
    for (int r = 0; r < 8; ++r) {
      int mm = r + sub * 8;
      int ng = nt * 16 + ln;
      float vv = c[r] + b1[ng] + se1[st[mm] * NUNIT + ng];
      sHa[mm * NUNIT + ng] = (_Float16)fmaxf(vv, 0.f);
    }
  }
  __syncthreads();

  // ---- layer 2: [16,128] @ W2[128,128] ----
  for (int nt = wave; nt < 8; nt += 4) {
    v8f c = {0.f, 0.f, 0.f, 0.f, 0.f, 0.f, 0.f, 0.f};
#pragma unroll
    for (int kc = 0; kc < 4; ++kc) {
      int kb = kc * 32;
      v8h lo = *(const v8h*)&sHa[ln * NUNIT + kb + sub * 8];
      v8h hi = *(const v8h*)&sHa[ln * NUNIT + kb + 16 + sub * 8];
      v16h af = cat16(lo, hi);
      v16h bf = *(const v16h*)(hp + OFF_W2 + ((nt * 4 + kc) * 32 + lane) * 16);
      c = wmma16(af, bf, c);
    }
#pragma unroll
    for (int r = 0; r < 8; ++r) {
      int mm = r + sub * 8;
      int ng = nt * 16 + ln;
      float vv = c[r] + b2[ng] + se2[st[mm] * NUNIT + ng];
      sHb[mm * NUNIT + ng] = (_Float16)fmaxf(vv, 0.f);
    }
  }
  __syncthreads();

  // ---- layer 3: [16,128] @ W3[128,128] ----
  for (int nt = wave; nt < 8; nt += 4) {
    v8f c = {0.f, 0.f, 0.f, 0.f, 0.f, 0.f, 0.f, 0.f};
#pragma unroll
    for (int kc = 0; kc < 4; ++kc) {
      int kb = kc * 32;
      v8h lo = *(const v8h*)&sHb[ln * NUNIT + kb + sub * 8];
      v8h hi = *(const v8h*)&sHb[ln * NUNIT + kb + 16 + sub * 8];
      v16h af = cat16(lo, hi);
      v16h bf = *(const v16h*)(hp + OFF_W3 + ((nt * 4 + kc) * 32 + lane) * 16);
      c = wmma16(af, bf, c);
    }
#pragma unroll
    for (int r = 0; r < 8; ++r) {
      int mm = r + sub * 8;
      int ng = nt * 16 + ln;
      float vv = c[r] + b3[ng] + se3[st[mm] * NUNIT + ng];
      sHa[mm * NUNIT + ng] = (_Float16)fmaxf(vv, 0.f);
    }
  }
  __syncthreads();

  // ---- layer 4: [16,128] @ W4[128,16] -> d_out columns 1..16 ----
  if (wave == 0) {
    v8f c = {0.f, 0.f, 0.f, 0.f, 0.f, 0.f, 0.f, 0.f};
#pragma unroll
    for (int kc = 0; kc < 4; ++kc) {
      int kb = kc * 32;
      v8h lo = *(const v8h*)&sHa[ln * NUNIT + kb + sub * 8];
      v8h hi = *(const v8h*)&sHa[ln * NUNIT + kb + 16 + sub * 8];
      v16h af = cat16(lo, hi);
      v16h bf = *(const v16h*)(hp + OFF_W4 + (kc * 32 + lane) * 16);
      c = wmma16(af, bf, c);
    }
#pragma unroll
    for (int r = 0; r < 8; ++r) {
      int mm = r + sub * 8;
      out[(long)(row0 + mm) * 17 + 1 + ln] = c[r] + b4[ln];
    }
  }
}

// ---------------- KGCN branch: one block per batch row, writes d_out[:,0] ----------------
__global__ void __launch_bounds__(256)
kgcn_kernel(const float* __restrict__ ent_emb, const float* __restrict__ rel_emb,
            const float* __restrict__ usr_emb, const float* __restrict__ agg_b,
            const float* __restrict__ fc1_W, const float* __restrict__ fc1_b,
            const float* __restrict__ fc2_W, const float* __restrict__ fc2_b,
            const int* __restrict__ u, const int* __restrict__ v,
            const int* __restrict__ click, const int* __restrict__ adj_ent,
            const int* __restrict__ adj_rel, const _Float16* __restrict__ hp,
            float* __restrict__ out) {
  __shared__ float    s_user[16];
  __shared__ float    s_urel[64];     // dot(user, rel_emb[j])/16 for all 64 relations
  __shared__ int      s_e0[64];
  __shared__ int      s_e1[512];
  __shared__ int      s_r0[512];
  __shared__ _Float16 s_A[576 * 16];  // WMMA A tiles (f16)
  __shared__ _Float16 s_H[576 * 16];  // sigmoid outputs of first pass
  __shared__ float    s_p0[512];      // cached hop-0 softmax, reused in iter 1
  __shared__ float    s_item[64 * 16];
  __shared__ float    s_fus[32];
  __shared__ float    s_fc1o[64];
  __shared__ float    s_feat[16];

  const int tid  = threadIdx.x;
  const int wave = tid >> 5;
  const int lane = tid & 31;
  const int sub  = lane >> 4;
  const int ln   = lane & 15;
  const int bb   = blockIdx.x;

  if (tid < 16) s_user[tid] = usr_emb[(long)u[bb] * ND + tid];
  if (tid >= 64 && tid < 128) s_e0[tid - 64] = click[bb * NS + (tid - 64)];
  __syncthreads();

  if (tid < 64) {
    float acc = 0.f;
#pragma unroll
    for (int d2 = 0; d2 < ND; ++d2) acc += s_user[d2] * rel_emb[tid * ND + d2];
    s_urel[tid] = acc * (1.0f / 16.0f);
  }
  for (int idx = tid; idx < 512; idx += 256) {
    int e = s_e0[idx >> 3];
    s_e1[idx] = adj_ent[(long)e * NKN + (idx & 7)];
    s_r0[idx] = adj_rel[(long)e * NKN + (idx & 7)];
  }
  __syncthreads();

  // ---- phase 1: build all 576 aggregate rows (64 s * (8 hop1 + 1 hop0)) ----
  for (int it = 0; it < 36; ++it) {
    int R = it * 16 + wave * 2 + sub;   // 0..575
    int s = R / 9, n = R - s * 9;
    int selfIdx;
    int ni[8];
    float sc[8];
    if (n < 8) {                        // hop-1 row n: self = e1[s][n], neigh = adj(e1)
      selfIdx = s_e1[s * 8 + n];
#pragma unroll
      for (int k = 0; k < 8; ++k) {
        ni[k] = adj_ent[(long)selfIdx * NKN + k];
        sc[k] = s_urel[adj_rel[(long)selfIdx * NKN + k]];
      }
    } else {                            // hop-0 row: self = e0[s], neigh = e1
      selfIdx = s_e0[s];
#pragma unroll
      for (int k = 0; k < 8; ++k) {
        ni[k] = s_e1[s * 8 + k];
        sc[k] = s_urel[s_r0[s * 8 + k]];
      }
    }
    float mx = sc[0];
#pragma unroll
    for (int k = 1; k < 8; ++k) mx = fmaxf(mx, sc[k]);
    float p[8], ssum = 0.f;
#pragma unroll
    for (int k = 0; k < 8; ++k) { p[k] = __expf(sc[k] - mx); ssum += p[k]; }
    float inv = 1.0f / ssum;
    float x = ent_emb[(long)selfIdx * ND + ln];
#pragma unroll
    for (int k = 0; k < 8; ++k) { p[k] *= inv; x += p[k] * ent_emb[(long)ni[k] * ND + ln]; }
    if (n == 8 && ln == 0) {
#pragma unroll
      for (int k = 0; k < 8; ++k) s_p0[s * 8 + k] = p[k];
    }
    s_A[R * 16 + ln] = (_Float16)x;
  }
  __syncthreads();

  // ---- B fragment of agg_W (pre-packed, K zero-padded to 32) ----
  const v16h bw = *(const v16h*)(hp + OFF_AGG + lane * 16);

  // ---- phase 2: 36 WMMA tiles, sigmoid ----
  for (int tile = wave; tile < 36; tile += 8) {
    v8h lo = *(const v8h*)&s_A[(tile * 16 + ln) * 16 + sub * 8];
    v16h af = cat16(lo, zero8());
    v8f c = {0.f, 0.f, 0.f, 0.f, 0.f, 0.f, 0.f, 0.f};
    c = wmma16(af, bw, c);
#pragma unroll
    for (int r = 0; r < 8; ++r) {
      int R = tile * 16 + r + sub * 8;
      float vv = c[r] + agg_b[ln];
      s_H[R * 16 + ln] = (_Float16)(1.0f / (1.0f + __expf(-vv)));
    }
  }
  __syncthreads();

  // ---- phase 3: iter-1 combine with cached p0, 4 WMMA tiles, tanh+relu ----
  for (int it = 0; it < 4; ++it) {
    int s = it * 16 + wave * 2 + sub;   // 0..63
    float x = (float)s_H[(s * 9 + 8) * 16 + ln];
#pragma unroll
    for (int k = 0; k < 8; ++k) x += s_p0[s * 8 + k] * (float)s_H[(s * 9 + k) * 16 + ln];
    s_A[s * 16 + ln] = (_Float16)x;
  }
  __syncthreads();
  if (wave < 4) {
    int tile = wave;
    v8h lo = *(const v8h*)&s_A[(tile * 16 + ln) * 16 + sub * 8];
    v16h af = cat16(lo, zero8());
    v8f c = {0.f, 0.f, 0.f, 0.f, 0.f, 0.f, 0.f, 0.f};
    c = wmma16(af, bw, c);
#pragma unroll
    for (int r = 0; r < 8; ++r) {
      int s = tile * 16 + r + sub * 8;
      float vv = tanhf(c[r] + agg_b[ln]);
      s_item[s * 16 + ln] = fmaxf(vv, 0.f);
    }
  }
  __syncthreads();

  // ---- phase 4: pooling + FC head ----
  if (tid < 16) {
    float ssum = 0.f, mx = -1e30f;
    for (int s = 0; s < 64; ++s) {
      float vv = s_item[s * 16 + tid];
      ssum += vv;
      mx = fmaxf(mx, vv);
    }
    s_fus[tid]      = fmaxf(ssum, 0.f);
    s_fus[16 + tid] = fmaxf(mx, 0.f);
  }
  __syncthreads();
  if (tid < 64) {
    float acc = fc1_b[tid];
#pragma unroll
    for (int i = 0; i < 32; ++i) acc += s_fus[i] * fc1_W[i * 64 + tid];
    s_fc1o[tid] = fmaxf(acc, 0.f);
  }
  __syncthreads();
  if (tid < 16) {
    float acc = fc2_b[tid];
    for (int j = 0; j < 64; ++j) acc += s_fc1o[j] * fc2_W[j * 16 + tid];
    s_feat[tid] = fmaxf(acc, 0.f);
  }
  __syncthreads();
  if (tid == 0) {
    long vb = v[bb];
    float acc = 0.f;
#pragma unroll
    for (int d2 = 0; d2 < ND; ++d2) acc += s_feat[d2] * ent_emb[vb * ND + d2];
    out[(long)bb * 17] = 1.0f / (1.0f + __expf(-acc));
  }
}

extern "C" void kernel_launch(void* const* d_in, const int* in_sizes, int n_in,
                              void* d_out, int out_size, void* d_ws, size_t ws_size,
                              hipStream_t stream) {
  const float* usr_emb = (const float*)d_in[0];
  const float* ent_emb = (const float*)d_in[1];
  const float* rel_emb = (const float*)d_in[2];
  const float* agg_W   = (const float*)d_in[3];
  const float* agg_b   = (const float*)d_in[4];
  const float* fc1_W   = (const float*)d_in[5];
  const float* fc1_b   = (const float*)d_in[6];
  const float* fc2_W   = (const float*)d_in[7];
  const float* fc2_b   = (const float*)d_in[8];
  const float* W1      = (const float*)d_in[9];
  const float* b1      = (const float*)d_in[10];
  const float* W2      = (const float*)d_in[11];
  const float* b2      = (const float*)d_in[12];
  const float* W3      = (const float*)d_in[13];
  const float* b3      = (const float*)d_in[14];
  const float* W4      = (const float*)d_in[15];
  const float* b4      = (const float*)d_in[16];
  const float* se1     = (const float*)d_in[17];
  const float* se2     = (const float*)d_in[18];
  const float* se3     = (const float*)d_in[19];
  const float* noise   = (const float*)d_in[20];
  const int*   u       = (const int*)d_in[21];
  const int*   v       = (const int*)d_in[22];
  const int*   click   = (const int*)d_in[23];
  const int*   adj_ent = (const int*)d_in[24];
  const int*   adj_rel = (const int*)d_in[25];
  const int*   t       = (const int*)d_in[26];
  float* out = (float*)d_out;

  float* ws_ab = (float*)d_ws;            // [100]
  float* ws_ob = ws_ab + NSTEP;           // [100]
  _Float16* h16 = (_Float16*)((char*)d_ws + 800);  // packed f16 weights (32B aligned)

  sched_kernel<<<1, 128, 0, stream>>>(ws_ab, ws_ob);
  prep_kernel<<<(PACK_TOT + 255) / 256, 256, 0, stream>>>(agg_W, W1, W2, W3, W4, h16);
  diff_kernel<<<NB / 16, 128, 0, stream>>>(usr_emb, noise, u, t, b1, b2, b3, b4,
                                           se1, se2, se3, h16, ws_ab, ws_ob, out);
  kgcn_kernel<<<NB, 256, 0, stream>>>(ent_emb, rel_emb, usr_emb, agg_b,
                                      fc1_W, fc1_b, fc2_W, fc2_b,
                                      u, v, click, adj_ent, adj_rel, h16, out);
}
